// PairRepresentationInit_58213986730476
// MI455X (gfx1250) — matrix operand order repaired and checked by reading, not compiled, not run on hardware
//
#include <hip/hip_runtime.h>
#include <hip/hip_bf16.h>

typedef __attribute__((ext_vector_type(16))) _Float16 v16h;
typedef __attribute__((ext_vector_type(8)))  float    v8f;

#define NRES  512
#define DTOK  256
#define DPAIR 128
#define DPROJ 32
#define KPAD  64

// workspace layout (bytes)
#define WS_W16T   0          // _Float16[128*64]  transposed fused weights W16T[n][k]
#define WS_BIAS   16384      // float[128]        dist_b + dir_b + outer_b
#define WS_LEFT   16896      // float[2*512*32]
#define WS_RIGHT  147968     // float[2*512*32]

// ---------------------------------------------------------------------------
// Kernel 0a: fuse [dist_W(16x128); dir_W(3x128); outer_W(32x128); zero-pad]
// into f16, stored transposed W16T[n*64+k] so B-fragments load contiguously.
// ---------------------------------------------------------------------------
__global__ void build_weights_kernel(const float* __restrict__ distW, const float* __restrict__ distB,
                                     const float* __restrict__ dirW,  const float* __restrict__ dirB,
                                     const float* __restrict__ outerW,const float* __restrict__ outerB,
                                     _Float16* __restrict__ w16t, float* __restrict__ bias) {
  int id = blockIdx.x * blockDim.x + threadIdx.x;   // 0..8191
  if (id < DPAIR * KPAD) {
    int n = id >> 6, k = id & 63;
    float v = 0.f;
    if (k < 16)      v = distW[k * DPAIR + n];
    else if (k < 19) v = dirW[(k - 16) * DPAIR + n];
    else if (k < 51) v = outerW[(k - 19) * DPAIR + n];
    w16t[n * KPAD + k] = (_Float16)v;
  }
  if (id < DPAIR) bias[id] = distB[id] + dirB[id] + outerB[id];
}

// ---------------------------------------------------------------------------
// Kernel 0b: left/right projections  token(256) @ W(256x32) + b
// grid.x = B*N rows, grid.y = 0(left)/1(right), 32 threads (one per output).
// ---------------------------------------------------------------------------
__global__ void proj_kernel(const float* __restrict__ tok,
                            const float* __restrict__ lW, const float* __restrict__ lb,
                            const float* __restrict__ rW, const float* __restrict__ rb,
                            float* __restrict__ left, float* __restrict__ right) {
  int row = blockIdx.x;
  int p   = threadIdx.x;
  const float* W = blockIdx.y ? rW : lW;
  float acc      = blockIdx.y ? rb[p] : lb[p];
  const float* trow = tok + (size_t)row * DTOK;
  for (int k = 0; k < DTOK; ++k) acc += trow[k] * W[k * DPROJ + p];
  (blockIdx.y ? right : left)[(size_t)row * DPROJ + p] = acc;
}

// ---------------------------------------------------------------------------
// Fused feature value: K rows 0..15 = rbf, 16..18 = unit dir, 19..50 = outer.
// ---------------------------------------------------------------------------
__device__ __forceinline__ float featval(int k, float d, float ux, float uy, float uz,
                                         const float* __restrict__ lft,
                                         const float* __restrict__ rrow) {
  if (k < 16) {
    float c = (20.0f / 15.0f) * (float)k;    // linspace(0,20,16)
    float z = (d - c) * 0.8f;                // sigma = 20/16 = 1.25
    return __expf(-z * z);
  }
  if (k < 19) return (k == 16) ? ux : ((k == 17) ? uy : uz);
  if (k < 51) { int p = k - 19; return lft[p] * rrow[p]; }
  return 0.f;
}

// ---------------------------------------------------------------------------
// Main kernel: one block per (b,i). 256 threads = 8 waves; each wave does
// 4 tiles of 16 j-values, each tile = 8 N-tiles x 2 K-steps of
// v_wmma_f32_16x16x32_f16 against the fused 64x128 weight matrix.
// Epilogue: table gathers issued before the WMMAs (latency hidden under the
// matrix op), all load/store addresses use compile-time immediate offsets.
// ---------------------------------------------------------------------------
__global__ __launch_bounds__(256)
void pair_main_kernel(const float* __restrict__ coords, const int* __restrict__ resIdx,
                      const unsigned char* __restrict__ seqMask,
                      const float* __restrict__ table,
                      const _Float16* __restrict__ w16t, const float* __restrict__ bias,
                      const float* __restrict__ left, const float* __restrict__ right,
                      float* __restrict__ out) {
  __shared__ float s_d[NRES], s_ux[NRES], s_uy[NRES], s_uz[NRES], s_mask[NRES];
  __shared__ int   s_rel[NRES];
  __shared__ float s_left[DPROJ];

  const int bi  = blockIdx.x;          // b*N + i
  const int b   = bi >> 9;
  const int i   = bi & (NRES - 1);
  const int tid = threadIdx.x;

  const size_t cai = ((size_t)bi * 14 + 1) * 3;
  const float cax = coords[cai + 0];
  const float cay = coords[cai + 1];
  const float caz = coords[cai + 2];
  const int   ri    = resIdx[bi];
  const float maski = seqMask[bi] ? 1.f : 0.f;

  // Per-block precompute: distance, unit vector, rel-pos row index, mask.
  for (int j = tid; j < NRES; j += 256) {
    size_t cj = ((size_t)(b * NRES + j) * 14 + 1) * 3;
    float dx = cax - coords[cj + 0];
    float dy = cay - coords[cj + 1];
    float dz = caz - coords[cj + 2];
    float d  = sqrtf(dx * dx + dy * dy + dz * dz);
    float inv = 1.f / (d + 1e-8f);
    bool diag = (j == i);
    s_d[j]  = d;
    s_ux[j] = diag ? 0.f : dx * inv;
    s_uy[j] = diag ? 0.f : dy * inv;
    s_uz[j] = diag ? 0.f : dz * inv;
    int r = ri - resIdx[b * NRES + j];
    r = (r < -32) ? -32 : ((r > 32) ? 32 : r);
    s_rel[j]  = (r + 32) * DPAIR;
    s_mask[j] = seqMask[b * NRES + j] ? 1.f : 0.f;
  }
  if (tid < DPROJ) s_left[tid] = left[(size_t)bi * DPROJ + tid];
  __syncthreads();

  const int lane = tid & 31;
  const int wv   = tid >> 5;
  const int hi   = lane >> 4;     // lane half
  const int m    = lane & 15;

  // B fragments: K x N = 64x128 fused weights, held in registers for the block.
  // Lane layout chosen here (contiguous k, same as A): n = n0 + (lane&15); element e -> k = 32*s + 16*hi + e.
  // W16T[n][k] is contiguous in k -> one 32-byte vector load per fragment.
  v16h Bf[8][2];
  #pragma unroll
  for (int t = 0; t < 8; ++t) {
    #pragma unroll
    for (int s = 0; s < 2; ++s) {
      int n = t * 16 + m;
      Bf[t][s] = *(const v16h*)(w16t + (size_t)n * KPAD + s * 32 + hi * 16);
    }
  }
  float biasv[8];
  #pragma unroll
  for (int t = 0; t < 8; ++t) biasv[t] = bias[t * 16 + m];

  for (int tile = wv; tile < 32; tile += 8) {
    const int j0 = tile * 16;
    const int j  = j0 + m;       // this lane's A-matrix row (pair index)
    const float d  = s_d[j];
    const float ux = s_ux[j], uy = s_uy[j], uz = s_uz[j];
    const float* rrow = right + (size_t)(b * NRES + j) * DPROJ;

    // A fragments (contiguous k order, same as B): element e -> k = 16*hi + e.
    v16h a0, a1;
    #pragma unroll
    for (int e = 0; e < 16; ++e) {
      int kb = hi * 16 + e;
      a0[e] = (_Float16)featval(kb,      d, ux, uy, uz, s_left, rrow);
      a1[e] = (_Float16)featval(kb + 32, d, ux, uy, uz, s_left, rrow);
    }

    // Per-output-row bases (C/D layout: row = r + 8*hi, col = m + 16*t).
    // All subsequent loads/stores use compile-time immediate offsets.
    const float* tb[8]; float mk[8];
    float* outp = out + ((size_t)bi * NRES + (size_t)(j0 + hi * 8)) * DPAIR + m;
    #pragma unroll
    for (int r = 0; r < 8; ++r) {
      int jj = j0 + r + hi * 8;
      tb[r] = table + s_rel[jj] + m;
      mk[r] = maski * s_mask[jj];
    }

    #pragma unroll
    for (int t = 0; t < 8; ++t) {
      // Issue the 8 table gathers first; the two WMMAs below execute while
      // these are in flight (also fills the WMMA->VALU hazard slots).
      float tv[8];
      #pragma unroll
      for (int r = 0; r < 8; ++r) tv[r] = tb[r][t * 16];

      v8f c = {};
      c = __builtin_amdgcn_wmma_f32_16x16x32_f16(false, a0, false, Bf[t][0],
                                                 (short)0, c, false, false);
      c = __builtin_amdgcn_wmma_f32_16x16x32_f16(false, a1, false, Bf[t][1],
                                                 (short)0, c, false, false);

      #pragma unroll
      for (int r = 0; r < 8; ++r) {
        float v = (c[r] + tv[r] + biasv[t]) * mk[r];
        __builtin_nontemporal_store(v, outp + r * DPAIR + t * 16);
      }
    }
  }
}

extern "C" void kernel_launch(void* const* d_in, const int* in_sizes, int n_in,
                              void* d_out, int out_size, void* d_ws, size_t ws_size,
                              hipStream_t stream) {
  const float* token  = (const float*)d_in[0];
  const float* coords = (const float*)d_in[1];
  const int*   resIdx = (const int*)d_in[2];
  const unsigned char* seqMask = (const unsigned char*)d_in[3];
  const float* table  = (const float*)d_in[4];
  const float* distW  = (const float*)d_in[5];
  const float* distB  = (const float*)d_in[6];
  const float* dirW   = (const float*)d_in[7];
  const float* dirB   = (const float*)d_in[8];
  const float* leftW  = (const float*)d_in[9];
  const float* leftB  = (const float*)d_in[10];
  const float* rightW = (const float*)d_in[11];
  const float* rightB = (const float*)d_in[12];
  const float* outerW = (const float*)d_in[13];
  const float* outerB = (const float*)d_in[14];

  char* ws = (char*)d_ws;
  _Float16* w16t = (_Float16*)(ws + WS_W16T);
  float* bias  = (float*)(ws + WS_BIAS);
  float* left  = (float*)(ws + WS_LEFT);
  float* right = (float*)(ws + WS_RIGHT);

  build_weights_kernel<<<32, 256, 0, stream>>>(distW, distB, dirW, dirB,
                                               outerW, outerB, w16t, bias);
  dim3 pg(2 * NRES, 2);
  proj_kernel<<<pg, DPROJ, 0, stream>>>(token, leftW, leftB, rightW, rightB, left, right);

  pair_main_kernel<<<2 * NRES, 256, 0, stream>>>(coords, resIdx, seqMask, table,
                                                 w16t, bias, left, right, (float*)d_out);
}
